// TFReformerLM_89275190215108
// MI455X (gfx1250) — matrix operand order repaired and checked by reading, not compiled, not run on hardware
//
#include <hip/hip_runtime.h>
#include <hip/hip_bf16.h>
#include <math.h>

// ---- problem constants (match reference) ----
#define S_LEN   4096
#define EMB     512
#define NHEAD   8
#define DHEAD   64
#define NHASH   4
#define NBUCK   64
#define VOCAB   32000
#define FFN     2048
#define DEPTH   2

typedef __attribute__((ext_vector_type(16))) __bf16 bf16x16;
typedef __attribute__((ext_vector_type(8)))  float  f32x8;

union FragU { bf16x16 v; uint4 q[2]; };

static __device__ __forceinline__ f32x8 zero8() {
  f32x8 z = {0.f, 0.f, 0.f, 0.f, 0.f, 0.f, 0.f, 0.f};
  return z;
}

// WMMA D = A(16x32 bf16) * B(32x16 bf16) + C(f32)
static __device__ __forceinline__ f32x8 wmma_bf16(bf16x16 a, bf16x16 b, f32x8 c) {
  return __builtin_amdgcn_wmma_f32_16x16x32_bf16(false, a, false, b, (short)0, c, false, false);
}

// A fragment: 16x32 tile, row-major bf16, stride ld (elements).
// lane (m=lane&15, half=lane>>4) holds row M=m, K = 16*half .. 16*half+15 contiguous (matches B)
static __device__ __forceinline__ bf16x16 load_frag_a(const __bf16* base, int ld) {
  int lane = threadIdx.x & 31;
  int hf = lane >> 4;
  int m  = lane & 15;
  const __bf16* p = base + (size_t)m * ld;
  FragU f;
  f.q[0] = *reinterpret_cast<const uint4*>(p + 16 * hf);
  f.q[1] = *reinterpret_cast<const uint4*>(p + 16 * hf + 8);
  return f.v;
}

// B fragment: B is 32x16 (KxN); we read from Bt (NxK row-major, stride ld):
// lane (n=lane&15, half=lane>>4) holds column n, K = 16*half .. 16*half+15 contiguous.
static __device__ __forceinline__ bf16x16 load_frag_b(const __bf16* base, int ld) {
  int lane = threadIdx.x & 31;
  int hf = lane >> 4;
  int n  = lane & 15;
  const __bf16* p = base + (size_t)n * ld + 16 * hf;
  FragU f;
  f.q[0] = *reinterpret_cast<const uint4*>(p);
  f.q[1] = *reinterpret_cast<const uint4*>(p + 8);
  return f.v;
}

// ---- CDNA5 async global->LDS (16 bytes per lane), tracked by ASYNCcnt ----
static __device__ __forceinline__ unsigned lds_off(const void* p) {
  // flat LDS aperture: addr[31:0] is the LDS byte offset
  return (unsigned)(unsigned long long)p;
}
static __device__ __forceinline__ void async_g2l_b128(unsigned lds_addr, const void* gaddr) {
  unsigned long long ga = (unsigned long long)gaddr;
  asm volatile("global_load_async_to_lds_b128 %0, %1, off"
               :: "v"(lds_addr), "v"(ga) : "memory");
}
static __device__ __forceinline__ void wait_async0() {
#if __has_builtin(__builtin_amdgcn_s_wait_asynccnt)
  __builtin_amdgcn_s_wait_asynccnt(0);
#else
  asm volatile("s_wait_asynccnt 0" ::: "memory");
#endif
}

static __device__ __forceinline__ float gelu_f(float x) {
  const float c = 0.7978845608028654f; // sqrt(2/pi)
  return 0.5f * x * (1.0f + tanhf(c * (x + 0.044715f * x * x * x)));
}

// ======================= embed =======================
__global__ __launch_bounds__(256) void embed_kernel(const int* __restrict__ tokens,
                                                    const float* __restrict__ tok_emb,
                                                    const float* __restrict__ pos_emb,
                                                    float* __restrict__ x1,
                                                    float* __restrict__ x2) {
  int idx = blockIdx.x * 256 + threadIdx.x;       // S*E total
  int t = idx >> 9;
  int e = idx & (EMB - 1);
  float v = tok_emb[(size_t)tokens[t] * EMB + e] + pos_emb[idx];
  x1[idx] = v;
  x2[idx] = v;
}

// ======================= layernorm -> bf16 =======================
__global__ __launch_bounds__(256) void layernorm_bf16(const float* __restrict__ x,
                                                      const float* __restrict__ g,
                                                      const float* __restrict__ b,
                                                      __bf16* __restrict__ out) {
  int row = blockIdx.x * 8 + (threadIdx.x >> 5);   // one wave per row
  int lane = threadIdx.x & 31;
  const float* xr = x + (size_t)row * EMB;
  float vals[16];
  float s = 0.f;
#pragma unroll
  for (int i = 0; i < 16; ++i) { vals[i] = xr[lane + i * 32]; s += vals[i]; }
#pragma unroll
  for (int m = 16; m; m >>= 1) s += __shfl_xor(s, m, 32);
  float mu = s * (1.0f / (float)EMB);
  float vs = 0.f;
#pragma unroll
  for (int i = 0; i < 16; ++i) { float d = vals[i] - mu; vs += d * d; }
#pragma unroll
  for (int m = 16; m; m >>= 1) vs += __shfl_xor(vs, m, 32);
  float rstd = rsqrtf(vs * (1.0f / (float)EMB) + 1e-3f);
#pragma unroll
  for (int i = 0; i < 16; ++i) {
    int e = lane + i * 32;
    out[(size_t)row * EMB + e] = (__bf16)((vals[i] - mu) * rstd * g[e] + b[e]);
  }
}

// ======================= transpose f32 -> bf16 (Bt for GEMM) =======================
__global__ __launch_bounds__(256) void transpose_bf(const float* __restrict__ in,
                                                    __bf16* __restrict__ out,
                                                    int R, int C) { // in[R][C] -> out[C][R]
  __shared__ float tile[32][33];
  int bc = blockIdx.x * 32;
  int br = blockIdx.y * 32;
  int tx = threadIdx.x & 31, ty = threadIdx.x >> 5;
  for (int i = ty; i < 32; i += 8)
    tile[i][tx] = in[(size_t)(br + i) * C + bc + tx];
  __syncthreads();
  for (int i = ty; i < 32; i += 8)
    out[(size_t)(bc + i) * R + br + tx] = (__bf16)tile[tx][i];
}

// ======================= bf16 WMMA GEMM =======================
// C[M][N] = A[M][K](bf16,rm) @ Bt[N][K](bf16,rm)^T  (+bias)(+gelu)(+add)
// Tiles staged into LDS with CDNA5 async global->LDS (ASYNCcnt), WMMAs fed by ds_load_b128.
#define BM 128
#define BN 64
#define BK 64
__global__ __launch_bounds__(256) void gemm_bf16(const __bf16* __restrict__ A,
                                                 const __bf16* __restrict__ Bt,
                                                 float* __restrict__ Cf,
                                                 __bf16* __restrict__ Cbf,
                                                 const float* __restrict__ bias,
                                                 const float* __restrict__ addsrc,
                                                 int M, int N, int K, int act) {
  __shared__ __bf16 As[BM][BK];  // 16 KB
  __shared__ __bf16 Bs[BN][BK];  //  8 KB
  int tid = threadIdx.x;
  int wave = tid >> 5;
  int lane = tid & 31;
  int hf = lane >> 4;
  int ln = lane & 15;
  int bm = blockIdx.y * BM;
  int bn = blockIdx.x * BN;
  int wm = (wave >> 1) * 32;   // 0,32,64,96
  int wn = (wave & 1) * 32;    // 0,32
  f32x8 acc[2][2];
#pragma unroll
  for (int i = 0; i < 2; ++i)
#pragma unroll
    for (int j = 0; j < 2; ++j) acc[i][j] = zero8();

  // A tile: 128 rows x 64; each thread moves 64B (row = tid/2, 32-elem segment)
  int arow = tid >> 1;
  int aseg = (tid & 1) * 32;
  unsigned lds_a = lds_off(&As[arow][aseg]);
  // Bt tile: 64 rows x 64; each thread moves 32B (row = tid/4, 16-elem segment)
  int brow = tid >> 2;
  int bseg = (tid & 3) * 16;
  unsigned lds_b = lds_off(&Bs[brow][bseg]);

  for (int k0 = 0; k0 < K; k0 += BK) {
    const __bf16* ga = A + (size_t)(bm + arow) * K + k0 + aseg;
    async_g2l_b128(lds_a +  0, ga +  0);
    async_g2l_b128(lds_a + 16, ga +  8);
    async_g2l_b128(lds_a + 32, ga + 16);
    async_g2l_b128(lds_a + 48, ga + 24);
    const __bf16* gb = Bt + (size_t)(bn + brow) * K + k0 + bseg;
    async_g2l_b128(lds_b +  0, gb +  0);
    async_g2l_b128(lds_b + 16, gb +  8);
    if (k0 + BK < K)
      __builtin_prefetch(A + (size_t)(bm + arow) * K + k0 + BK + aseg, 0, 1);
    wait_async0();
    __syncthreads();

#pragma unroll
    for (int kk = 0; kk < BK; kk += 32) {
      bf16x16 a0 = load_frag_a(&As[wm + 0][kk], BK);
      bf16x16 a1 = load_frag_a(&As[wm + 16][kk], BK);
      bf16x16 b0 = load_frag_b(&Bs[wn + 0][kk], BK);
      bf16x16 b1 = load_frag_b(&Bs[wn + 16][kk], BK);
      acc[0][0] = wmma_bf16(a0, b0, acc[0][0]);
      acc[0][1] = wmma_bf16(a0, b1, acc[0][1]);
      acc[1][0] = wmma_bf16(a1, b0, acc[1][0]);
      acc[1][1] = wmma_bf16(a1, b1, acc[1][1]);
    }
    __syncthreads();
  }
  // epilogue: C layout row = reg + 8*half, col = lane&15
#pragma unroll
  for (int i = 0; i < 2; ++i) {
#pragma unroll
    for (int j = 0; j < 2; ++j) {
      int gn = bn + wn + j * 16 + ln;
      float bv = bias ? bias[gn] : 0.0f;
#pragma unroll
      for (int rg = 0; rg < 8; ++rg) {
        int gm = bm + wm + i * 16 + rg + 8 * hf;
        float vv = acc[i][j][rg] + bv;
        if (act) vv = gelu_f(vv);
        size_t o = (size_t)gm * (size_t)N + (size_t)gn;
        if (addsrc) vv += addsrc[o];
        if (Cf) Cf[o] = vv;
        if (Cbf) Cbf[o] = (__bf16)vv;
      }
    }
  }
}

// ======================= LSH hash =======================
// keys[(h*NHASH+r)*S + t] = (argmax over [rot,-rot] << 12) | t
__global__ __launch_bounds__(256) void lsh_hash(const float* __restrict__ qk,
                                                const float* __restrict__ rot,
                                                unsigned* __restrict__ keys) {
  int idx = blockIdx.x * 256 + threadIdx.x;       // NHEAD*NHASH*S
  int t = idx & (S_LEN - 1);
  int r = (idx >> 12) & 3;
  int h = idx >> 14;
  const float* q = qk + (size_t)t * EMB + h * DHEAD;
  float vals[32];
#pragma unroll
  for (int i = 0; i < 32; ++i) vals[i] = 0.f;
  for (int d = 0; d < DHEAD; ++d) {
    float qd = q[d];
    const float* rp = rot + ((size_t)d * NHASH + r) * 32;
#pragma unroll
    for (int i = 0; i < 32; ++i) vals[i] += qd * rp[i];
  }
  int best = 0;
  float bv = vals[0];
#pragma unroll
  for (int i = 1; i < 32; ++i) if (vals[i] > bv) { bv = vals[i]; best = i; }
#pragma unroll
  for (int i = 0; i < 32; ++i) if (-vals[i] > bv) { bv = -vals[i]; best = i + 32; }
  keys[((size_t)h * NHASH + r) * S_LEN + t] = ((unsigned)best << 12) | (unsigned)t;
}

// ======================= bitonic sort (per head,round) =======================
__global__ __launch_bounds__(256) void bitonic_sort(unsigned* __restrict__ keys) {
  __shared__ unsigned sk[S_LEN];
  unsigned* g = keys + (size_t)blockIdx.x * S_LEN;
  int tid = threadIdx.x;
  for (int i = tid; i < S_LEN; i += 256) sk[i] = g[i];
  __syncthreads();
  for (int k = 2; k <= S_LEN; k <<= 1) {
    for (int j = k >> 1; j > 0; j >>= 1) {
      for (int i = tid; i < S_LEN; i += 256) {
        int ixj = i ^ j;
        if (ixj > i) {
          bool up = ((i & k) == 0);
          unsigned a = sk[i], b = sk[ixj];
          bool sw = up ? (a > b) : (a < b);
          if (sw) { sk[i] = b; sk[ixj] = a; }
        }
      }
      __syncthreads();
    }
  }
  for (int i = tid; i < S_LEN; i += 256) g[i] = sk[i];
}

// ======================= chunked LSH attention =======================
struct alignas(16) AttnLDS {
  __bf16 Qb[64][64];     // bq (gathered, unnormalized)
  __bf16 Kb[128][64];    // bk (normalized, current + prev chunk)
  __bf16 Vt[64][128];    // V^T (d-major) for P@V B-operand
  __bf16 Pb[64][128];    // softmax probabilities
  float  dots[64][128];
  int    qt[64];  int qb[64];
  int    kt[128]; int kb[128];
  float  kscale[128];
  float  lsev[64];
};

__global__ __launch_bounds__(256) void lsh_attn(const float* __restrict__ qk,
                                                const float* __restrict__ v,
                                                const unsigned* __restrict__ keys,
                                                float* __restrict__ o_r,
                                                float* __restrict__ lse_r) {
  extern __shared__ char smem_raw[];
  AttnLDS& L = *reinterpret_cast<AttnLDS*>(smem_raw);
  int tid = threadIdx.x;
  int blk = blockIdx.x;
  int h  = blk >> 8;        // head
  int cc = blk & 255;       // global chunk (NHASH*NBUCK), crosses rounds on look-back
  int r = cc >> 6, c = cc & 63;
  int ccp = (cc + 255) & 255;
  int rp = ccp >> 6, cp = ccp & 63;
  const unsigned* kcur  = keys + ((size_t)h * NHASH + r)  * S_LEN;
  const unsigned* kprev = keys + ((size_t)h * NHASH + rp) * S_LEN;

  if (tid < 64) {
    unsigned key = kcur[c * 64 + tid];
    L.qt[tid] = (int)(key & 4095u);
    L.qb[tid] = (int)(key >> 12) + r * NBUCK;   // global bucket id
  } else if (tid < 192) {
    int jj = tid - 64;
    unsigned key;
    int roff;
    if (jj < 64) { key = kcur[c * 64 + jj];        roff = r * NBUCK; }
    else         { key = kprev[cp * 64 + (jj-64)]; roff = rp * NBUCK; }
    L.kt[jj] = (int)(key & 4095u);
    L.kb[jj] = (int)(key >> 12) + roff;
  }
  __syncthreads();

  if (tid < 128) { // key row scales: 1/max(||k||, 1e-9)
    const float* row = qk + (size_t)L.kt[tid] * EMB + h * DHEAD;
    float ss = 0.f;
    for (int d = 0; d < DHEAD; ++d) { float x = row[d]; ss += x * x; }
    L.kscale[tid] = 1.0f / fmaxf(sqrtf(ss), 1e-9f);
  }
  __syncthreads();

  for (int idx = tid; idx < 64 * 64; idx += 256) {   // gather Q
    int i = idx >> 6, d = idx & 63;
    L.Qb[i][d] = (__bf16)qk[(size_t)L.qt[i] * EMB + h * DHEAD + d];
  }
  for (int idx = tid; idx < 128 * 64; idx += 256) {  // gather K (normalized) and V^T
    int jj = idx >> 6, d = idx & 63;
    int t = L.kt[jj];
    L.Kb[jj][d] = (__bf16)(qk[(size_t)t * EMB + h * DHEAD + d] * L.kscale[jj]);
    L.Vt[d][jj] = (__bf16)v[(size_t)t * EMB + h * DHEAD + d];
  }
  __syncthreads();

  int wave = tid >> 5, lane = tid & 31;
  int hf = lane >> 4, ln = lane & 15;
  int wm  = (wave >> 1) * 16;   // query row base (4 waves over 64 rows)
  {
    // dots[64][128] = Q @ K^T * d^-0.5, masked
    int wn0 = (wave & 1) * 64;  // 4 column tiles per wave
    f32x8 acc[4];
#pragma unroll
    for (int j = 0; j < 4; ++j) acc[j] = zero8();
    for (int k0 = 0; k0 < 64; k0 += 32) {
      bf16x16 a = load_frag_a(&L.Qb[wm][k0], 64);
#pragma unroll
      for (int j = 0; j < 4; ++j) {
        bf16x16 b = load_frag_b(&L.Kb[wn0 + j * 16][k0], 64);
        acc[j] = wmma_bf16(a, b, acc[j]);
      }
    }
#pragma unroll
    for (int j = 0; j < 4; ++j) {
      int col = wn0 + j * 16 + ln;
      int ktj = L.kt[col], kbj = L.kb[col];
#pragma unroll
      for (int rg = 0; rg < 8; ++rg) {
        int row = wm + rg + 8 * hf;
        float dv = acc[j][rg] * 0.125f;       // 1/sqrt(64)
        int qtr = L.qt[row];
        if (qtr < ktj)            dv = -1e9f; // causal
        if (qtr == ktj)           dv = -1e5f; // self
        if (L.qb[row] != kbj)     dv = -1e7f; // cross-bucket (applied last)
        L.dots[row][col] = dv;
      }
    }
  }
  __syncthreads();

  if (tid < 64) { // row softmax with lse
    float mx = -INFINITY;
    for (int j = 0; j < 128; ++j) mx = fmaxf(mx, L.dots[tid][j]);
    float s = 0.f;
    for (int j = 0; j < 128; ++j) s += expf(L.dots[tid][j] - mx);
    float lse = logf(s) + mx;
    L.lsev[tid] = lse;
    for (int j = 0; j < 128; ++j) L.Pb[tid][j] = (__bf16)expf(L.dots[tid][j] - lse);
    lse_r[((size_t)h * NHASH + r) * S_LEN + L.qt[tid]] = lse;
  }
  __syncthreads();

  { // out[64][64] = P @ V, scatter to o_r by original position
    int wn = (wave & 1) * 32;
    f32x8 acc[2];
#pragma unroll
    for (int j = 0; j < 2; ++j) acc[j] = zero8();
    for (int k0 = 0; k0 < 128; k0 += 32) {
      bf16x16 a = load_frag_a(&L.Pb[wm][k0], 128);
#pragma unroll
      for (int j = 0; j < 2; ++j) {
        bf16x16 b = load_frag_b(&L.Vt[wn + j * 16][k0], 128);
        acc[j] = wmma_bf16(a, b, acc[j]);
      }
    }
    float* obase = o_r + ((size_t)h * NHASH + r) * S_LEN * DHEAD;
#pragma unroll
    for (int j = 0; j < 2; ++j) {
      int col = wn + j * 16 + ln;
#pragma unroll
      for (int rg = 0; rg < 8; ++rg) {
        int row = wm + rg + 8 * hf;
        obase[(size_t)L.qt[row] * DHEAD + col] = acc[j][rg];
      }
    }
  }
}

// ======================= combine hash rounds =======================
__global__ __launch_bounds__(256) void combine_rounds(const float* __restrict__ o_r,
                                                      const float* __restrict__ lse_r,
                                                      __bf16* __restrict__ attn_bf) {
  int idx = blockIdx.x * 256 + threadIdx.x;  // NHEAD * S
  int h = idx >> 12;
  int t = idx & (S_LEN - 1);
  float l[NHASH];
  float mx = -INFINITY;
#pragma unroll
  for (int r = 0; r < NHASH; ++r) {
    l[r] = lse_r[((size_t)h * NHASH + r) * S_LEN + t];
    mx = fmaxf(mx, l[r]);
  }
  float s = 0.f;
#pragma unroll
  for (int r = 0; r < NHASH; ++r) { l[r] = expf(l[r] - mx); s += l[r]; }
  float inv = 1.0f / s;
  for (int d = 0; d < DHEAD; ++d) {
    float o = 0.f;
#pragma unroll
    for (int r = 0; r < NHASH; ++r)
      o += o_r[(((size_t)h * NHASH + r) * S_LEN + t) * DHEAD + d] * (l[r] * inv);
    attn_bf[(size_t)t * EMB + h * DHEAD + d] = (__bf16)o;
  }
}

// ======================= final y = x1 + x2 (bf16) =======================
__global__ __launch_bounds__(256) void add_bf16(const float* __restrict__ a,
                                                const float* __restrict__ b,
                                                __bf16* __restrict__ out) {
  int i = blockIdx.x * 256 + threadIdx.x;
  out[i] = (__bf16)(a[i] + b[i]);
}

// ======================= host orchestration =======================
extern "C" void kernel_launch(void* const* d_in, const int* in_sizes, int n_in,
                              void* d_out, int out_size, void* d_ws, size_t ws_size,
                              hipStream_t stream) {
  (void)in_sizes; (void)n_in; (void)out_size; (void)ws_size;
  const int*   tokens  = (const int*)d_in[0];
  const float* tok_emb = (const float*)d_in[1];
  const float* pos_emb = (const float*)d_in[2];
  const float* ln1_g   = (const float*)d_in[3];
  const float* ln1_b   = (const float*)d_in[4];
  const float* Wqk     = (const float*)d_in[5];
  const float* Wv      = (const float*)d_in[6];
  const float* Wo      = (const float*)d_in[7];
  const float* b_o     = (const float*)d_in[8];
  const float* ln2_g   = (const float*)d_in[9];
  const float* ln2_b   = (const float*)d_in[10];
  const float* W1      = (const float*)d_in[11];
  const float* b1      = (const float*)d_in[12];
  const float* W2      = (const float*)d_in[13];
  const float* b2      = (const float*)d_in[14];
  const float* rot     = (const float*)d_in[15];
  const float* Wlog    = (const float*)d_in[16];
  const float* blog    = (const float*)d_in[17];
  float* out = (float*)d_out;

  char* ws = (char*)d_ws;
  size_t off = 0;
  auto carve = [&](size_t bytes) -> void* {
    void* p = ws + off;
    off += (bytes + 255) & ~(size_t)255;
    return p;
  };
  float*    x1      = (float*)carve((size_t)S_LEN * EMB * 4);
  float*    x2      = (float*)carve((size_t)S_LEN * EMB * 4);
  __bf16*   xn_bf   = (__bf16*)carve((size_t)S_LEN * EMB * 2);
  float*    qk      = (float*)carve((size_t)S_LEN * EMB * 4);
  float*    vbuf    = (float*)carve((size_t)S_LEN * EMB * 4);
  unsigned* keys    = (unsigned*)carve((size_t)NHEAD * NHASH * S_LEN * 4);
  float*    o_r     = (float*)carve((size_t)NHEAD * NHASH * S_LEN * DHEAD * 4);
  float*    lse_r   = (float*)carve((size_t)NHEAD * NHASH * S_LEN * 4);
  __bf16*   attn_bf = (__bf16*)carve((size_t)S_LEN * EMB * 2);
  __bf16*   h_bf    = (__bf16*)carve((size_t)S_LEN * FFN * 2);
  __bf16*   y_bf    = (__bf16*)carve((size_t)S_LEN * EMB * 2);
  __bf16*   wt_qk   = (__bf16*)carve((size_t)EMB * EMB * 2);
  __bf16*   wt_v    = (__bf16*)carve((size_t)EMB * EMB * 2);
  __bf16*   wt_o    = (__bf16*)carve((size_t)EMB * EMB * 2);
  __bf16*   wt_1    = (__bf16*)carve((size_t)EMB * FFN * 2);
  __bf16*   wt_2    = (__bf16*)carve((size_t)FFN * EMB * 2);
  __bf16*   wt_log  = (__bf16*)carve((size_t)VOCAB * EMB * 2);

  // embed (x1 = x2 = tok_emb[tokens] + pos_emb)
  embed_kernel<<<(S_LEN * EMB) / 256, 256, 0, stream>>>(tokens, tok_emb, pos_emb, x1, x2);

  // logits weight transpose once
  transpose_bf<<<dim3(VOCAB / 32, EMB / 32), 256, 0, stream>>>(Wlog, wt_log, EMB, VOCAB);

  for (int layer = 0; layer < DEPTH; ++layer) {
    const float* Wqk_i = Wqk + (size_t)layer * EMB * EMB;
    const float* Wv_i  = Wv  + (size_t)layer * EMB * EMB;
    const float* Wo_i  = Wo  + (size_t)layer * EMB * EMB;
    const float* W1_i  = W1  + (size_t)layer * EMB * FFN;
    const float* W2_i  = W2  + (size_t)layer * FFN * EMB;
    const float* rot_i = rot + (size_t)layer * DHEAD * NHASH * (NBUCK / 2);

    transpose_bf<<<dim3(EMB / 32, EMB / 32), 256, 0, stream>>>(Wqk_i, wt_qk, EMB, EMB);
    transpose_bf<<<dim3(EMB / 32, EMB / 32), 256, 0, stream>>>(Wv_i,  wt_v,  EMB, EMB);
    transpose_bf<<<dim3(EMB / 32, EMB / 32), 256, 0, stream>>>(Wo_i,  wt_o,  EMB, EMB);
    transpose_bf<<<dim3(FFN / 32, EMB / 32), 256, 0, stream>>>(W1_i,  wt_1,  EMB, FFN);
    transpose_bf<<<dim3(EMB / 32, FFN / 32), 256, 0, stream>>>(W2_i,  wt_2,  FFN, EMB);

    // f-branch: attn(LN(x2)) added to x1
    layernorm_bf16<<<S_LEN / 8, 256, 0, stream>>>(x2, ln1_g + layer * EMB, ln1_b + layer * EMB, xn_bf);
    gemm_bf16<<<dim3(EMB / BN, S_LEN / BM), 256, 0, stream>>>(
        xn_bf, wt_qk, qk, nullptr, nullptr, nullptr, S_LEN, EMB, EMB, 0);
    gemm_bf16<<<dim3(EMB / BN, S_LEN / BM), 256, 0, stream>>>(
        xn_bf, wt_v, vbuf, nullptr, nullptr, nullptr, S_LEN, EMB, EMB, 0);

    lsh_hash<<<(NHEAD * NHASH * S_LEN) / 256, 256, 0, stream>>>(qk, rot_i, keys);
    bitonic_sort<<<NHEAD * NHASH, 256, 0, stream>>>(keys);
    lsh_attn<<<NHEAD * NHASH * NBUCK, 256, sizeof(AttnLDS), stream>>>(qk, vbuf, keys, o_r, lse_r);
    combine_rounds<<<(NHEAD * S_LEN) / 256, 256, 0, stream>>>(o_r, lse_r, attn_bf);

    // y1 = x1 + attn @ Wo + b_o   (in-place into x1)
    gemm_bf16<<<dim3(EMB / BN, S_LEN / BM), 256, 0, stream>>>(
        attn_bf, wt_o, x1, nullptr, b_o + layer * EMB, x1, S_LEN, EMB, EMB, 0);

    // g-branch: FFN(LN(y1)) added to x2
    layernorm_bf16<<<S_LEN / 8, 256, 0, stream>>>(x1, ln2_g + layer * EMB, ln2_b + layer * EMB, xn_bf);
    gemm_bf16<<<dim3(FFN / BN, S_LEN / BM), 256, 0, stream>>>(
        xn_bf, wt_1, nullptr, h_bf, b1 + layer * FFN, nullptr, S_LEN, FFN, EMB, 1);
    gemm_bf16<<<dim3(EMB / BN, S_LEN / BM), 256, 0, stream>>>(
        h_bf, wt_2, x2, nullptr, b2 + layer * EMB, x2, S_LEN, EMB, FFN, 0);
  }

  // y = x1 + x2 ; logits = y @ Wlog + blog
  add_bf16<<<(S_LEN * EMB) / 256, 256, 0, stream>>>(x1, x2, y_bf);
  gemm_bf16<<<dim3(VOCAB / BN, S_LEN / BM), 256, 0, stream>>>(
      y_bf, wt_log, out, nullptr, blog, nullptr, S_LEN, VOCAB, EMB, 0);
}
